// MTA_sharemoe4_58634893525404
// MI455X (gfx1250) — compile-verified, matrix operand order checked by reading, not run on hardware
//
#include <hip/hip_runtime.h>

// ---------------- types ----------------
typedef __bf16 bf16;
typedef bf16  v16bf __attribute__((ext_vector_type(16)));
typedef bf16  v8bf  __attribute__((ext_vector_type(8)));
typedef float v8f   __attribute__((ext_vector_type(8)));

// problem constants (match reference)
constexpr int Bc = 8, Sc = 1024, Dc = 1024, Fc = 4096;

// ---------------- f32 -> bf16 straight convert (vector x8) ----------------
__global__ void __launch_bounds__(256)
convert_f32_bf16(const float* __restrict__ in, bf16* __restrict__ out, size_t n) {
  size_t i = ((size_t)blockIdx.x * 256 + threadIdx.x) * 8;
  if (i >= n) return;
  float4 a = *(const float4*)(in + i);
  float4 b = *(const float4*)(in + i + 4);
  v8bf o;
  o[0] = (bf16)a.x; o[1] = (bf16)a.y; o[2] = (bf16)a.z; o[3] = (bf16)a.w;
  o[4] = (bf16)b.x; o[5] = (bf16)b.y; o[6] = (bf16)b.z; o[7] = (bf16)b.w;
  *(v8bf*)(out + i) = o;
}

// ---------------- f32 [R][C] -> bf16 [C][R] transpose+convert ----------------
// grid: (C/32, R/32, numMatrices), block (32,8)
__global__ void __launch_bounds__(256)
convT_f32_bf16(const float* __restrict__ src, bf16* __restrict__ dst, int R, int C) {
  __shared__ float tile[32][33];
  size_t mat = (size_t)blockIdx.z * R * C;
  const float* s = src + mat;
  bf16* d = dst + mat;
  int c0 = blockIdx.x * 32, r0 = blockIdx.y * 32;
#pragma unroll
  for (int i = 0; i < 4; ++i) {
    int rr = threadIdx.y + i * 8;
    tile[rr][threadIdx.x] = s[(size_t)(r0 + rr) * C + (c0 + threadIdx.x)];
  }
  __syncthreads();
#pragma unroll
  for (int i = 0; i < 4; ++i) {
    int cc = threadIdx.y + i * 8;
    d[(size_t)(c0 + cc) * R + (r0 + threadIdx.x)] = (bf16)tile[threadIdx.x][cc];
  }
}

// ---------------- WMMA GEMM ----------------
// C[b] = (relu?)( scale * A[b] @ W(b)^T )  where W is stored transposed [N][K]
// RELU_BF16: 1 -> relu + bf16 output (Cb), 0 -> f32 output (Cf) with scale/beta.
// Routed expert selection: W += (type_idx[b]+expert_offset)*wstride.
// If gate_vecs != nullptr, scale = softmax(gate_vecs[t]/0.1)[t+expert_offset].
template <int RELU_BF16>
__global__ void __launch_bounds__(256)
gemm_wmma(const bf16* __restrict__ A,      // [B][M][K]
          const bf16* __restrict__ W,      // [N][K] (maybe per-expert base)
          size_t wstride,
          const int* __restrict__ type_idx,
          int expert_offset,
          const float* __restrict__ gate_vecs,
          float* __restrict__ Cf,
          bf16* __restrict__ Cb,
          int M, int N, int K, int beta) {
  const int b  = blockIdx.z;
  const int n0 = blockIdx.x * 128;
  const int m0 = blockIdx.y * 128;

  const bf16* Ab = A + (size_t)b * M * K;
  const bf16* Wb = W + (type_idx ? (size_t)(type_idx[b] + expert_offset) * wstride : 0);

  float scale = 1.0f;
  if (gate_vecs) {
    int t = type_idx[b];
    float v[4], mx = -1e30f, s = 0.f;
#pragma unroll
    for (int i = 0; i < 4; ++i) { v[i] = gate_vecs[t * 4 + i] * 10.0f; mx = fmaxf(mx, v[i]); }
#pragma unroll
    for (int i = 0; i < 4; ++i) { v[i] = expf(v[i] - mx); s += v[i]; }
    scale = v[t + expert_offset] / s;
  }

  // LDS tiles in fragment-permuted layout: [tile16][lane32][16 bf16]
  __shared__ __attribute__((aligned(32))) bf16 As[2][4096];
  __shared__ __attribute__((aligned(32))) bf16 Bs[2][4096];

  const int tid  = threadIdx.x;
  const int lane = tid & 31;
  const int wave = tid >> 5;
  const int wm   = wave >> 2;  // 0..1 -> 64 rows each
  const int wn   = wave & 3;   // 0..3 -> 32 cols each

  v8f acc[4][2];
#pragma unroll
  for (int i = 0; i < 4; ++i)
#pragma unroll
    for (int j = 0; j < 2; ++j)
#pragma unroll
      for (int r = 0; r < 8; ++r) acc[i][j][r] = 0.f;

  const int nk = K >> 5;  // BK = 32

  // async tile copy: global -> LDS directly (ASYNCcnt), no VGPR staging
  auto load_tile_async = [&](int kb, int buf) {
#pragma unroll
    for (int rep = 0; rep < 2; ++rep) {
      int p   = tid + rep * 256;       // 0..511 -> (row 0..127, chunk 0..3)
      int row = p >> 2;
      int t4  = p & 3;                 // 8-element K chunk
      int g   = t4 & 1;                // lane half (K group)
      int eb  = (t4 >> 1) * 8;         // fragment element base
      int off = (row >> 4) * 512 + ((row & 15) + 16 * g) * 16 + eb;
      // low 32 bits of a flat LDS pointer == LDS byte address (aperture mapping)
      unsigned lA = (unsigned)(uintptr_t)(&As[buf][off]);
      unsigned lB = (unsigned)(uintptr_t)(&Bs[buf][off]);
      unsigned long long gA =
          (unsigned long long)(uintptr_t)(Ab + (size_t)(m0 + row) * K + (size_t)kb * 32 + t4 * 8);
      unsigned long long gB =
          (unsigned long long)(uintptr_t)(Wb + (size_t)(n0 + row) * K + (size_t)kb * 32 + t4 * 8);
      asm volatile("global_load_async_to_lds_b128 %0, %1, off" ::"v"(lA), "v"(gA) : "memory");
      asm volatile("global_load_async_to_lds_b128 %0, %1, off" ::"v"(lB), "v"(gB) : "memory");
    }
  };

  load_tile_async(0, 0);
  asm volatile("s_wait_asynccnt 0x0" ::: "memory");
  __syncthreads();

  for (int kb = 0; kb < nk; ++kb) {
    int buf = kb & 1;
    if (kb + 1 < nk) load_tile_async(kb + 1, buf ^ 1);

    v16bf afr[4], bfr[2];
#pragma unroll
    for (int i = 0; i < 4; ++i)
      afr[i] = *(const v16bf*)(&As[buf][(wm * 4 + i) * 512 + lane * 16]);
#pragma unroll
    for (int j = 0; j < 2; ++j)
      bfr[j] = *(const v16bf*)(&Bs[buf][(wn * 2 + j) * 512 + lane * 16]);

#pragma unroll
    for (int i = 0; i < 4; ++i)
#pragma unroll
      for (int j = 0; j < 2; ++j)
        acc[i][j] = __builtin_amdgcn_wmma_f32_16x16x32_bf16(
            false, afr[i], false, bfr[j], (short)0, acc[i][j], false, false);

    asm volatile("s_wait_asynccnt 0x0" ::: "memory");
    __syncthreads();
  }

  // epilogue: C 16x16 f32 layout -> lane(0..15)=N, VGPR r: M=r (+8 for lanes 16..31)
  const int lcol = lane & 15;
  const int lrow = (lane >> 4) * 8;
#pragma unroll
  for (int i = 0; i < 4; ++i) {
#pragma unroll
    for (int j = 0; j < 2; ++j) {
      int col = n0 + wn * 32 + j * 16 + lcol;
#pragma unroll
      for (int r = 0; r < 8; ++r) {
        int row = m0 + wm * 64 + i * 16 + lrow + r;
        size_t off = (size_t)b * M * N + (size_t)row * N + col;
        float val = acc[i][j][r] * scale;
        if (RELU_BF16) {
          Cb[off] = (bf16)fmaxf(val, 0.f);
        } else {
          Cf[off] = beta ? (Cf[off] + val) : val;
        }
      }
    }
  }
}

// ---------------- gate MLP stage 1: t1 = relu(gi @ W1), gi = concat of token-0 rows ----------------
// grid (D/64, B), block 64
__global__ void __launch_bounds__(64)
gate1_kernel(const float* __restrict__ eo, const float* __restrict__ s0,
             const float* __restrict__ s1, const float* __restrict__ s2,
             const float* __restrict__ w1, float* __restrict__ t1) {
  int j = blockIdx.x * 64 + threadIdx.x;
  int b = blockIdx.y;
  const float* parts[4] = {eo + (size_t)b * Sc * Dc, s0 + (size_t)b * Sc * Dc,
                           s1 + (size_t)b * Sc * Dc, s2 + (size_t)b * Sc * Dc};
  float acc = 0.f;
  for (int p = 0; p < 4; ++p) {
    const float* x = parts[p];
    const float* wrow = w1 + (size_t)p * Dc * Dc + j;
    for (int i = 0; i < Dc; ++i) acc = fmaf(x[i], wrow[(size_t)i * Dc], acc);
  }
  t1[b * Dc + j] = fmaxf(acc, 0.f);
}

// ---------------- gate MLP stage 2: g = softmax(t1 @ W2) ----------------
__global__ void __launch_bounds__(256)
gate2_kernel(const float* __restrict__ t1, const float* __restrict__ w2,
             float* __restrict__ g) {
  int b = blockIdx.x;
  __shared__ float red[256];
  __shared__ float z[4];
  float a[4] = {0.f, 0.f, 0.f, 0.f};
  for (int j = threadIdx.x; j < Dc; j += 256) {
    float h = t1[b * Dc + j];
#pragma unroll
    for (int c = 0; c < 4; ++c) a[c] = fmaf(h, w2[j * 4 + c], a[c]);
  }
  for (int c = 0; c < 4; ++c) {
    red[threadIdx.x] = a[c];
    __syncthreads();
    for (int s = 128; s > 0; s >>= 1) {
      if (threadIdx.x < s) red[threadIdx.x] += red[threadIdx.x + s];
      __syncthreads();
    }
    if (threadIdx.x == 0) z[c] = red[0];
    __syncthreads();
  }
  if (threadIdx.x == 0) {
    float mx = fmaxf(fmaxf(z[0], z[1]), fmaxf(z[2], z[3]));
    float e0 = expf(z[0] - mx), e1 = expf(z[1] - mx), e2 = expf(z[2] - mx), e3 = expf(z[3] - mx);
    float inv = 1.f / (e0 + e1 + e2 + e3);
    g[b * 4 + 0] = e0 * inv; g[b * 4 + 1] = e1 * inv;
    g[b * 4 + 2] = e2 * inv; g[b * 4 + 3] = e3 * inv;
  }
}

// ---------------- final 4-way weighted combine ----------------
__global__ void __launch_bounds__(256)
combine_kernel(const float4* __restrict__ eo, const float4* __restrict__ s0,
               const float4* __restrict__ s1, const float4* __restrict__ s2,
               const float* __restrict__ g, float4* __restrict__ out, int n4) {
  int i = blockIdx.x * 256 + threadIdx.x;
  if (i >= n4) return;
  int b = i >> 18;  // (i*4) / (S*D) with S*D = 2^20
  float g0 = g[b * 4 + 0], g1 = g[b * 4 + 1], g2 = g[b * 4 + 2], g3 = g[b * 4 + 3];
  float4 e = eo[i], x0 = s0[i], x1 = s1[i], x2 = s2[i];
  float4 o;
  o.x = e.x * g0 + x0.x * g1 + x1.x * g2 + x2.x * g3;
  o.y = e.y * g0 + x0.y * g1 + x1.y * g2 + x2.y * g3;
  o.z = e.z * g0 + x0.z * g1 + x1.z * g2 + x2.z * g3;
  o.w = e.w * g0 + x0.w * g1 + x1.w * g2 + x2.w * g3;
  out[i] = o;
}

// ---------------- launcher ----------------
extern "C" void kernel_launch(void* const* d_in, const int* in_sizes, int n_in,
                              void* d_out, int out_size, void* d_ws, size_t ws_size,
                              hipStream_t stream) {
  const float* hidden   = (const float*)d_in[0];
  const int*   type_idx = (const int*)d_in[1];
  const float* e_wi     = (const float*)d_in[2];   // [E][D][F]
  const float* e_wo     = (const float*)d_in[3];   // [E][F][D]
  const float* s_wi     = (const float*)d_in[4];   // [3][D][F]
  const float* s_wo     = (const float*)d_in[5];   // [3][F][D]
  const float* gvec     = (const float*)d_in[6];   // [3][4]
  const float* g_w1     = (const float*)d_in[7];   // [4D][D]
  const float* g_w2     = (const float*)d_in[8];   // [D][4]
  float* out = (float*)d_out;

  // workspace carve (~336 MB total)
  char* p = (char*)d_ws;
  auto carve = [&](size_t bytes) {
    char* q = p;
    p += (bytes + 255) & ~(size_t)255;
    return q;
  };
  bf16*  Xb   = (bf16*)carve((size_t)Bc * Sc * Dc * 2);
  bf16*  WiEt = (bf16*)carve((size_t)4 * Fc * Dc * 2);   // [E][F][D]
  bf16*  WoEt = (bf16*)carve((size_t)4 * Dc * Fc * 2);   // [E][D][F]
  bf16*  WiSt = (bf16*)carve((size_t)3 * Fc * Dc * 2);   // [3][F][D]
  bf16*  WoSt = (bf16*)carve((size_t)3 * Dc * Fc * 2);   // [3][D][F]
  bf16*  H    = (bf16*)carve((size_t)Bc * Sc * Fc * 2);  // reused hidden MLP act
  float* s0   = (float*)carve((size_t)Bc * Sc * Dc * 4);
  float* s1   = (float*)carve((size_t)Bc * Sc * Dc * 4);
  float* s2   = (float*)carve((size_t)Bc * Sc * Dc * 4);
  float* eo   = (float*)carve((size_t)Bc * Sc * Dc * 4);
  float* t1   = (float*)carve((size_t)Bc * Dc * 4);
  float* g    = (float*)carve((size_t)Bc * 4 * 4);

  // 1) bf16 conversions (weights transposed so GEMM contraction dim is contiguous)
  convert_f32_bf16<<<(Bc * Sc * Dc / 8 + 255) / 256, 256, 0, stream>>>(
      hidden, Xb, (size_t)Bc * Sc * Dc);
  convT_f32_bf16<<<dim3(Fc / 32, Dc / 32, 4), dim3(32, 8), 0, stream>>>(e_wi, WiEt, Dc, Fc);
  convT_f32_bf16<<<dim3(Dc / 32, Fc / 32, 4), dim3(32, 8), 0, stream>>>(e_wo, WoEt, Fc, Dc);
  convT_f32_bf16<<<dim3(Fc / 32, Dc / 32, 3), dim3(32, 8), 0, stream>>>(s_wi, WiSt, Dc, Fc);
  convT_f32_bf16<<<dim3(Dc / 32, Fc / 32, 3), dim3(32, 8), 0, stream>>>(s_wo, WoSt, Fc, Dc);

  dim3 g1grid(Fc / 128, Sc / 128, Bc);  // GEMM1: M=S, N=F, K=D
  dim3 g2grid(Dc / 128, Sc / 128, Bc);  // GEMM2: M=S, N=D, K=F

  // 2) shared experts
  float* souts[3] = {s0, s1, s2};
  for (int k = 0; k < 3; ++k) {
    gemm_wmma<1><<<g1grid, 256, 0, stream>>>(Xb, WiSt + (size_t)k * Fc * Dc, 0, nullptr, 0,
                                             nullptr, nullptr, H, Sc, Fc, Dc, 0);
    gemm_wmma<0><<<g2grid, 256, 0, stream>>>(H, WoSt + (size_t)k * Dc * Fc, 0, nullptr, 0,
                                             nullptr, souts[k], nullptr, Sc, Dc, Fc, 0);
  }

  // 3) routed experts: pass A overwrites eo with wa*y, pass B accumulates wb*y
  for (int which = 0; which < 2; ++which) {
    gemm_wmma<1><<<g1grid, 256, 0, stream>>>(Xb, WiEt, (size_t)Fc * Dc, type_idx, which,
                                             nullptr, nullptr, H, Sc, Fc, Dc, 0);
    gemm_wmma<0><<<g2grid, 256, 0, stream>>>(H, WoEt, (size_t)Dc * Fc, type_idx, which,
                                             gvec, eo, nullptr, Sc, Dc, Fc, which);
  }

  // 4) learned gate on token-0 features, then combine
  gate1_kernel<<<dim3(Dc / 64, Bc), 64, 0, stream>>>(eo, s0, s1, s2, g_w1, t1);
  gate2_kernel<<<Bc, 256, 0, stream>>>(t1, g_w2, g);
  combine_kernel<<<(Bc * Sc * Dc / 4) / 256, 256, 0, stream>>>(
      (const float4*)eo, (const float4*)s0, (const float4*)s1, (const float4*)s2, g,
      (float4*)out, Bc * Sc * Dc / 4);
}
